// CategoricalGraphAtt_60911226192237
// MI455X (gfx1250) — matrix operand order repaired and checked by reading, not compiled, not run on hardware
//
#include <hip/hip_runtime.h>
#include <hip/hip_bf16.h>
#include <hip/hip_fp16.h>

typedef _Float16 f16;
typedef __attribute__((ext_vector_type(16))) _Float16 v16h;
typedef __attribute__((ext_vector_type(8)))  float    v8f;

#define WN   4
#define NB   4000
#define TT   20
#define DINV 16
#define HH   128
#define GG   384     // 3*H
#define CC   10
#define DTOT 26
#define BPC  400     // nodes per category
#define KPAD 416     // 400 padded to multiple of 32

__device__ __forceinline__ float sigmoidf_(float x){ return 1.0f/(1.0f+__expf(-x)); }
__device__ __forceinline__ float lrelu_(float x){ return x>0.0f ? x : 0.2f*x; }

// A fragment (16xK tile, 16-bit): lane m=l&15, lane-half selects K sub-range,
// VGPR i holds packed pair (see CDNA5 ISA 7.12.2, 16-bit A 16x32 layout).
__device__ __forceinline__ v16h load_A_frag(const f16* base, int ld, int k0, int lane){
  const int m  = lane & 15;
  const int hi = (lane >> 4) & 1;
  const f16* p = base + m*ld + k0 + hi*8;
  v16h a;
#pragma unroll
  for (int i=0;i<8;++i){
    int kb = ((i>=4)?16:0) + (i&3)*2;
    a[2*i]   = p[kb];
    a[2*i+1] = p[kb+1];
  }
  return a;
}

// B fragment (Kx16 tile, 16-bit) from W stored [n][k] row-major (k contiguous):
// lane n=l&15, lane-half h: elems 0-7 -> k=8h+i, elems 8-15 -> k=16+8h+(i-8).
__device__ __forceinline__ v16h load_B_frag(const f16* Wnk, int ld, int n0, int k0, int lane){
  const f16* p = Wnk + (size_t)(n0 + (lane & 15))*ld + k0 + ((lane>>4)&1)*8;
  v16h b;
#pragma unroll
  for (int i=0;i<16;++i) b[i] = p[(i<8) ? i : i+8];
  return b;
}

// D fragment (16x16 f32): lane n=l&15, VGPR i -> row (i + 8*lanehalf).
__device__ __forceinline__ void store_D_lds(float* out, int ld, v8f d, int lane){
  const int n  = lane & 15;
  const int mb = ((lane >> 4) & 1) * 8;
#pragma unroll
  for (int i=0;i<8;++i) out[(mb+i)*ld + n] = d[i];
}

__device__ __forceinline__ v8f wmma16(v16h a, v16h b, v8f c){
  return __builtin_amdgcn_wmma_f32_16x16x32_f16(false, a, false, b, (short)0, c, false, false);
}

// ---------------------------------------------------------------------------
// Kernel A: per (week, 16-row tile) GRU over T steps + temporal attention.
// W_hh/W_ih/x-tile/hidden history all resident in LDS; WMMA f16->f32.
// ---------------------------------------------------------------------------
__global__ __launch_bounds__(256) void k_gru_att(
    const float* __restrict__ x,     // (W,N,T,26)
    const float* __restrict__ Wih,   // (W,384,16)
    const float* __restrict__ Whh,   // (W,384,128)
    const float* __restrict__ bih,   // (W,384)
    const float* __restrict__ bhh,   // (W,384)
    const float* __restrict__ attW,  // (W,20,20)
    const float* __restrict__ attB,  // (W,20)
    float* __restrict__ weekly_t)    // (N,W,128)
{
  __shared__ f16   sWih[GG*32];        // B layout [n][k], k padded 16->32 w/ zeros
  __shared__ f16   sWhh[GG*HH];        // B layout [n][k]
  __shared__ float sBih[GG], sBhh[GG];
  __shared__ f16   sX[16*TT*32];       // [m][t][k] k padded 16->32
  __shared__ f16   sH[16*HH];          // current hidden (A layout row-major)
  __shared__ f16   sHs[16*TT*HH];      // hidden history for attention
  __shared__ float sGi[16*GG], sGh[16*GG];

  const int w   = blockIdx.y;
  const int m0  = blockIdx.x * 16;
  const int tid = threadIdx.x;

  const float* wih = Wih + (size_t)w*GG*DINV;
  for (int i=tid;i<GG*32;i+=256){
    int n=i>>5, k=i&31;
    sWih[i] = (k<DINV) ? (f16)wih[n*DINV+k] : (f16)0.0f;
  }
  const float* whh = Whh + (size_t)w*GG*HH;
  for (int i=tid;i<GG*HH;i+=256) sWhh[i] = (f16)whh[i];
  for (int i=tid;i<GG;i+=256){ sBih[i]=bih[w*GG+i]; sBhh[i]=bhh[w*GG+i]; }
  for (int i=tid;i<16*TT*32;i+=256){
    int m=i/(TT*32), r=i%(TT*32), t=r>>5, k=r&31;
    sX[i] = (k<DINV)
      ? (f16)x[(((size_t)w*NB + (m0+m))*TT + t)*DTOT + k] : (f16)0.0f;
  }
  for (int i=tid;i<16*HH;i+=256) sH[i]=(f16)0.0f;
  __syncthreads();

  const int wave = tid>>5, lane = tid&31;

  for (int t=0;t<TT;++t){
#pragma unroll
    for (int q=0;q<3;++q){
      const int n0 = (wave*3+q)*16;
      v8f gi = {};
      {
        v16h a = load_A_frag(sX + t*32, TT*32, 0, lane);
        v16h b = load_B_frag(sWih, 32, n0, 0, lane);
        gi = wmma16(a,b,gi);
      }
      v8f gh = {};
#pragma unroll
      for (int kc=0;kc<4;++kc){
        v16h a = load_A_frag(sH, HH, kc*32, lane);
        v16h b = load_B_frag(sWhh, HH, n0, kc*32, lane);
        gh = wmma16(a,b,gh);
      }
      store_D_lds(sGi + n0, GG, gi, lane);
      store_D_lds(sGh + n0, GG, gh, lane);
    }
    __syncthreads();
#pragma unroll
    for (int r=0;r<8;++r){
      int idx = tid + 256*r;
      int m = idx>>7, j = idx&127;
      float ir = sGi[m*GG + j]        + sBih[j];
      float iz = sGi[m*GG + HH + j]   + sBih[HH+j];
      float in_= sGi[m*GG + 2*HH + j] + sBih[2*HH+j];
      float hr = sGh[m*GG + j]        + sBhh[j];
      float hz = sGh[m*GG + HH + j]   + sBhh[HH+j];
      float hn = sGh[m*GG + 2*HH + j] + sBhh[2*HH+j];
      float rg = sigmoidf_(ir+hr);
      float zg = sigmoidf_(iz+hz);
      float ng = tanhf(in_ + rg*hn);
      float hp = (float)sH[m*HH+j];
      float hv = (1.0f-zg)*ng + zg*hp;
      sH[m*HH+j]            = (f16)hv;
      sHs[(m*TT+t)*HH + j]  = (f16)hv;
    }
    __syncthreads();
  }

  // temporal attention: per (row,d) softmax over s of <hs[:,d], attW[s,:]>+b[s]
  const float* aw = attW + w*TT*TT;
  const float* ab = attB + w*TT;
#pragma unroll
  for (int r=0;r<8;++r){
    int idx = tid + 256*r;
    int m = idx>>7, d = idx&127;
    float lg[TT]; float mx = -1e30f;
    for (int s=0;s<TT;++s){
      float acc = ab[s];
      for (int t2=0;t2<TT;++t2)
        acc += (float)sHs[(m*TT+t2)*HH + d] * aw[s*TT+t2];
      lg[s]=acc; if (acc>mx) mx=acc;
    }
    float ss=0.0f;
    for (int s=0;s<TT;++s){ lg[s]=__expf(lg[s]-mx); ss+=lg[s]; }
    float o=0.0f;
    for (int s=0;s<TT;++s) o += lg[s]*(float)sHs[(m*TT+s)*HH + d];
    weekly_t[((size_t)(m0+m)*WN + w)*HH + d] = o/ss;
  }
}

// ---------------------------------------------------------------------------
// Kernel B: attention over the 4 weeks (elementwise per (row,d)).
// ---------------------------------------------------------------------------
__global__ void k_week_att(const float* __restrict__ weekly_t,
                           const float* __restrict__ Wk, const float* __restrict__ bk,
                           float* __restrict__ att_vec){
  int idx = blockIdx.x*blockDim.x + threadIdx.x;
  if (idx >= NB*HH) return;
  int b = idx>>7, d = idx&127;
  const float* base = weekly_t + (size_t)b*WN*HH + d;
  float v[WN];
#pragma unroll
  for (int t=0;t<WN;++t) v[t]=base[t*HH];
  float lg[WN]; float mx=-1e30f;
#pragma unroll
  for (int s=0;s<WN;++s){
    float a=bk[s];
#pragma unroll
    for (int t=0;t<WN;++t) a += v[t]*Wk[s*WN+t];
    lg[s]=a; if (a>mx) mx=a;
  }
  float ss=0.0f, o=0.0f;
#pragma unroll
  for (int s=0;s<WN;++s){ float e=__expf(lg[s]-mx); ss+=e; o+=e*v[s]; }
  att_vec[idx]=o/ss;
}

// ---------------------------------------------------------------------------
// Kernel C1: h = att_vec @ inner_W.T (WMMA) + per-node a_src/a_dst dots.
// ---------------------------------------------------------------------------
__global__ __launch_bounds__(256) void k_inner_gemm(
    const float* __restrict__ att_vec, const float* __restrict__ Wi,
    const float* __restrict__ a_src, const float* __restrict__ a_dst,
    float* __restrict__ hout, f16* __restrict__ hout16,
    float* __restrict__ as_, float* __restrict__ ad_)
{
  __shared__ f16   sA[16*HH];
  __shared__ f16   sB[HH*HH];
  __shared__ float sO[16*HH];
  const int m0=blockIdx.x*16, tid=threadIdx.x;
  for (int i=tid;i<16*HH;i+=256) sA[i]=(f16)att_vec[(size_t)m0*HH + i];
  for (int i=tid;i<HH*HH;i+=256) sB[i]=(f16)Wi[i];
  __syncthreads();
  const int wave=tid>>5, lane=tid&31, n0=wave*16;
  v8f acc={};
#pragma unroll
  for (int kc=0;kc<4;++kc){
    v16h a=load_A_frag(sA,HH,kc*32,lane);
    v16h b=load_B_frag(sB,HH,n0,kc*32,lane);
    acc=wmma16(a,b,acc);
  }
  store_D_lds(sO+n0,HH,acc,lane);
  __syncthreads();
  for (int i=tid;i<16*HH;i+=256){
    float v=sO[i];
    hout  [(size_t)m0*HH+i]=v;
    hout16[(size_t)m0*HH+i]=(f16)v;
  }
  if (tid<32){
    int m=tid&15;
    const float* av=(tid<16)?a_src:a_dst;
    float s=0.0f;
    for (int d=0;d<HH;++d) s += sO[m*HH+d]*av[d];
    if (tid<16) as_[m0+m]=s; else ad_[m0+m]=s;
  }
}

// ---------------------------------------------------------------------------
// Kernel C2: inner GAT. Clique structure: dst j receives src i with i<=j
// (local index within category). Per dst-tile: masked softmax row (f16 in LDS)
// then (16x416)@(416x128) WMMA against category feature block.
// ---------------------------------------------------------------------------
__global__ __launch_bounds__(256) void k_inner_gat(
    const f16* __restrict__ h16, const float* __restrict__ as_,
    const float* __restrict__ ad_, const float* __restrict__ bias,
    float* __restrict__ out)
{
  __shared__ f16   sHT[HH*KPAD];   // [d][k]  (B layout: n=d, k=src contiguous)
  __shared__ f16   sAl[16*KPAD];   // alpha tile
  __shared__ float sAs[BPC], sAd[BPC];
  __shared__ float sM[16], sS[16];
  const int c=blockIdx.x, tid=threadIdx.x, wave=tid>>5, lane=tid&31;
  const int base=c*BPC;
  for (int i=tid;i<HH*KPAD;i+=256){
    int d=i/KPAD, k=i%KPAD;
    sHT[i] = (k<BPC) ? h16[(size_t)(base+k)*HH + d] : (f16)0.0f;
  }
  for (int i=tid;i<BPC;i+=256){ sAs[i]=as_[base+i]; sAd[i]=ad_[base+i]; }
  __syncthreads();

  for (int dt=0; dt<BPC/16; ++dt){
    if (tid<16){
      int j = dt*16+tid;
      float adj = sAd[j];
      float mx=-1e30f, ss=0.0f;
      for (int i=0;i<=j;++i){
        float e=lrelu_(sAs[i]+adj);
        if (e>mx){ ss = ss*__expf(mx-e) + 1.0f; mx=e; }
        else        ss += __expf(e-mx);
      }
      sM[tid]=mx; sS[tid]=ss;
    }
    __syncthreads();
    for (int i=tid;i<16*KPAD;i+=256){
      int r=i/KPAD, k=i%KPAD;
      int j=dt*16+r;
      float v=0.0f;
      if (k<BPC && k<=j) v = __expf(lrelu_(sAs[k]+sAd[j]) - sM[r]) / sS[r];
      sAl[i]=(f16)v;
    }
    __syncthreads();
    const int n0=wave*16;
    v8f acc={};
#pragma unroll
    for (int kc=0;kc<KPAD/32;++kc){
      v16h a=load_A_frag(sAl,KPAD,kc*32,lane);
      v16h b=load_B_frag(sHT,KPAD,n0,kc*32,lane);
      acc=wmma16(a,b,acc);
    }
    {
      int n=n0+(lane&15); int mb=((lane>>4)&1)*8;
      float bn=bias[n];
#pragma unroll
      for (int i=0;i<8;++i){
        int row = base + dt*16 + mb + i;
        out[(size_t)row*HH + n] = acc[i] + bn;
      }
    }
    __syncthreads();
  }
}

// ---------------------------------------------------------------------------
// Kernel D: relu(segment_max) over each 400-node category.
// ---------------------------------------------------------------------------
__global__ void k_catmax(const float* __restrict__ inner, float* __restrict__ catv){
  int c=blockIdx.x, d=threadIdx.x;
  float mx=-1e30f;
  for (int i=0;i<BPC;++i){
    float v=inner[(size_t)(c*BPC+i)*HH + d];
    mx = v>mx ? v : mx;
  }
  catv[c*HH+d] = mx>0.0f ? mx : 0.0f;
}

// ---------------------------------------------------------------------------
// Kernel E: outer GAT on the 10 category nodes (tiny; VALU).
// ---------------------------------------------------------------------------
__global__ __launch_bounds__(256) void k_outer_gat(
    const float* __restrict__ catv, const float* __restrict__ Wc,
    const float* __restrict__ a_src, const float* __restrict__ a_dst,
    const float* __restrict__ bias, float* __restrict__ cat_out)
{
  __shared__ float sV[CC*HH];
  __shared__ float sHc[CC*HH];
  __shared__ float sAs[CC], sAd[CC];
  const int tid=threadIdx.x;
  for (int i=tid;i<CC*HH;i+=256) sV[i]=catv[i];
  __syncthreads();
  for (int i=tid;i<CC*HH;i+=256){
    int j=i/HH, d=i%HH;
    float s=0.0f;
    for (int k=0;k<HH;++k) s+=sV[j*HH+k]*Wc[d*HH+k];
    sHc[i]=s;
  }
  __syncthreads();
  if (tid<2*CC){
    int j=tid%CC;
    const float* av=(tid<CC)?a_src:a_dst;
    float s=0.0f;
    for (int d=0;d<HH;++d) s+=sHc[j*HH+d]*av[d];
    if (tid<CC) sAs[j]=s; else sAd[j]=s;
  }
  __syncthreads();
  for (int i=tid;i<CC*HH;i+=256){
    int j=i/HH, d=i%HH;
    float mx=-1e30f, ss=0.0f;
    for (int q=0;q<=j;++q){
      float e=lrelu_(sAs[q]+sAd[j]);
      if (e>mx){ ss=ss*__expf(mx-e)+1.0f; mx=e; } else ss+=__expf(e-mx);
    }
    float o=0.0f;
    for (int q=0;q<=j;++q)
      o += (__expf(lrelu_(sAs[q]+sAd[j])-mx)/ss) * sHc[q*HH+d];
    cat_out[i] = o + bias[d];
  }
}

// ---------------------------------------------------------------------------
// Kernel F: fusion relu([att|inner|expand] @ Wf.T + b) via WMMA, then heads.
// ---------------------------------------------------------------------------
__global__ __launch_bounds__(256) void k_fusion(
    const float* __restrict__ att_vec, const float* __restrict__ inner,
    const float* __restrict__ cat_out, const int* __restrict__ catidx,
    const float* __restrict__ Wf, const float* __restrict__ bf,
    const float* __restrict__ regW, const float* __restrict__ regB,
    const float* __restrict__ clsW, const float* __restrict__ clsB,
    float* __restrict__ outp)
{
  __shared__ f16   sA[16*GG];
  __shared__ f16   sW[HH*GG];
  __shared__ float sO[16*HH];
  const int m0=blockIdx.x*16, tid=threadIdx.x;
  for (int i=tid;i<16*GG;i+=256){
    int m=i/GG, k=i%GG;
    int row=m0+m;
    float v;
    if      (k<HH)    v=att_vec[(size_t)row*HH + k];
    else if (k<2*HH)  v=inner  [(size_t)row*HH + (k-HH)];
    else              v=cat_out[catidx[row]*HH + (k-2*HH)];
    sA[i]=(f16)v;
  }
  for (int i=tid;i<HH*GG;i+=256) sW[i]=(f16)Wf[i];
  __syncthreads();
  const int wave=tid>>5, lane=tid&31, n0=wave*16;
  v8f acc={};
#pragma unroll
  for (int kc=0;kc<GG/32;++kc){
    v16h a=load_A_frag(sA,GG,kc*32,lane);
    v16h b=load_B_frag(sW,GG,n0,kc*32,lane);
    acc=wmma16(a,b,acc);
  }
  {
    int n=n0+(lane&15); int mb=((lane>>4)&1)*8;
    float bn=bf[n];
#pragma unroll
    for (int i=0;i<8;++i){
      float v=acc[i]+bn;
      sO[(mb+i)*HH + n] = v>0.0f ? v : 0.0f;
    }
  }
  __syncthreads();
  if (tid<16){
    float r=regB[0], cl=clsB[0];
    for (int d=0;d<HH;++d){
      float f=sO[tid*HH+d];
      r  += f*regW[d];
      cl += f*clsW[d];
    }
    outp[m0+tid]      = r;
    outp[NB+m0+tid]   = sigmoidf_(cl);
  }
}

// ---------------------------------------------------------------------------
extern "C" void kernel_launch(void* const* d_in, const int* in_sizes, int n_in,
                              void* d_out, int out_size, void* d_ws, size_t ws_size,
                              hipStream_t stream)
{
  (void)in_sizes; (void)n_in; (void)out_size; (void)ws_size;
  const float* x        = (const float*)d_in[0];
  const float* Wih      = (const float*)d_in[1];
  const float* Whh      = (const float*)d_in[2];
  const float* bih      = (const float*)d_in[3];
  const float* bhh      = (const float*)d_in[4];
  const float* attW     = (const float*)d_in[5];
  const float* attB     = (const float*)d_in[6];
  const float* wkW      = (const float*)d_in[7];
  const float* wkB      = (const float*)d_in[8];
  const float* innerW   = (const float*)d_in[9];
  const float* ia_src   = (const float*)d_in[10];
  const float* ia_dst   = (const float*)d_in[11];
  const float* i_bias   = (const float*)d_in[12];
  const float* catW     = (const float*)d_in[13];
  const float* ca_src   = (const float*)d_in[14];
  const float* ca_dst   = (const float*)d_in[15];
  const float* c_bias   = (const float*)d_in[16];
  const float* fusW     = (const float*)d_in[17];
  const float* fusB     = (const float*)d_in[18];
  const float* regW     = (const float*)d_in[19];
  const float* regB     = (const float*)d_in[20];
  const float* clsW     = (const float*)d_in[21];
  const float* clsB     = (const float*)d_in[22];
  const int*   catidx   = (const int*)d_in[23];

  float* ws = (float*)d_ws;
  float* weekly_t = ws;                  // N*W*H      = 2,048,000
  float* att_vec  = ws + 2048000;        // N*H        =   512,000
  float* h_inner  = ws + 2560000;        // N*H        =   512,000
  f16*   h16      = (f16*)(ws + 3072000);// N*H halves (256,000 floats)
  float* as_      = ws + 3328000;        // 4000
  float* ad_      = ws + 3332000;        // 4000
  float* inner_o  = ws + 3336000;        // N*H        =   512,000
  float* catv     = ws + 3848000;        // 1280
  float* cato     = ws + 3849280;        // 1280

  dim3 gA(NB/16, WN);
  k_gru_att<<<gA, 256, 0, stream>>>(x, Wih, Whh, bih, bhh, attW, attB, weekly_t);
  k_week_att<<<(NB*HH+255)/256, 256, 0, stream>>>(weekly_t, wkW, wkB, att_vec);
  k_inner_gemm<<<NB/16, 256, 0, stream>>>(att_vec, innerW, ia_src, ia_dst,
                                          h_inner, h16, as_, ad_);
  k_inner_gat<<<CC, 256, 0, stream>>>(h16, as_, ad_, i_bias, inner_o);
  k_catmax<<<CC, HH, 0, stream>>>(inner_o, catv);
  k_outer_gat<<<1, 256, 0, stream>>>(catv, catW, ca_src, ca_dst, c_bias, cato);
  k_fusion<<<NB/16, 256, 0, stream>>>(att_vec, inner_o, cato, catidx,
                                      fusW, fusB, regW, regB, clsW, clsB,
                                      (float*)d_out);
}
